// MoEFFN_59665685676633
// MI455X (gfx1250) — matrix operand order repaired and checked by reading, not compiled, not run on hardware
//
#include <hip/hip_runtime.h>
#include <hip/hip_bf16.h>
#include <cstdint>

// ---------------- problem constants (from reference) ----------------
#define T_TOK   8192          // B*S
#define DDIM    1024
#define HDIM    2048
#define NEXP    8
#define TOPK    2
#define CAP     1280          // int(1.25 * 8192 / 8)
#define MAXTOK  (2*CAP)       // per-expert compacted list upper bound
#define TILES   (MAXTOK/16)   // 160 16-token tiles per expert

typedef __attribute__((ext_vector_type(16))) __bf16 v16bf;
typedef __attribute__((ext_vector_type(8)))  float  v8f;

// Workspace layout: routing scratch in the first 512KB, then (optionally)
// bf16 fragment-packed weights (3 x 32MB). Packed path requires ~96.5MB ws.
#define ROUTE_BYTES   (512u * 1024u)
#define PACK_ELEMS    ((size_t)NEXP * HDIM * DDIM)          // per matrix
#define WS_REQUIRED   (ROUTE_BYTES + 3u * PACK_ELEMS * 2u)  // bytes

// ---------------- WMMA fragment loaders (wave32, 16x16x32 bf16) ------
// A (16x32, M x K): lane m(0..15)/m+16 holds row m with K runs
// [h*8 .. h*8+7] and [16+h*8 .. 16+h*8+7], h = lane>>4.
__device__ __forceinline__ v16bf load_a_bf16(const __bf16* base, int m, int h,
                                             int kb, int stride) {
  v16bf a;
  const __bf16* r = base + m * stride + kb + h * 8;
  #pragma unroll
  for (int i = 0; i < 8; ++i) { a[i] = r[i]; a[8 + i] = r[16 + i]; }
  return a;
}

// B (32x16, K x N): lane l holds N=l&15; K runs [h*8..h*8+7], [16+h*8..16+h*8+7].
// Fallback path: gather fp32 rows (coalesced across lanes) + cvt to bf16.
__device__ __forceinline__ v16bf load_b_f32(const float* __restrict__ W,
                                            int nbase, int kb, int stride,
                                            int lane) {
  v16bf b;
  const float* p = W + (size_t)(nbase + (lane & 15)) * stride + kb + (lane >> 4) * 8;
  #pragma unroll
  for (int i = 0; i < 8; ++i) { b[i] = (__bf16)p[i]; b[8 + i] = (__bf16)p[16 + i]; }
  return b;
}

// Packed path: fragment `f` is 1KB contiguous; lane reads 32B at f*512+lane*16
// (two global_load_b128, 1KB contiguous per wave -> perfectly coalesced).
__device__ __forceinline__ v16bf load_b_packed(const __bf16* __restrict__ P,
                                               int frag, int lane) {
  return *(const v16bf*)(P + (size_t)frag * 512 + lane * 16);
}

// ---------------- kernel 0: zero output --------------------------------
__global__ void zero_out_kernel(float* __restrict__ p, size_t n) {
  size_t i = (size_t)blockIdx.x * blockDim.x + threadIdx.x;
  size_t stride = (size_t)gridDim.x * blockDim.x;
  for (; i < n; i += stride) p[i] = 0.f;
}

// ---------------- kernel 0b: pack fp32 weights -> bf16 B-fragments ------
// Source W: (E, Nrows, Kcols) row-major. Dest fragment layout:
// P[((e*NT + nt)*KT + kt)*512 + lane*16 + i] = (bf16) W[e][nt*16+(lane&15)][kt*32+k(lane>>4,i)] (table order)
// One thread per (fragment, lane): 16 strided f32 reads (coalesced across
// lanes along K), one 32B contiguous bf16 vector write.
__global__ void pack_weights_kernel(const float* __restrict__ W,
                                    __bf16* __restrict__ P,
                                    int Nrows, int Kcols) {
  const int KT = Kcols / 32, NT = Nrows / 16;
  const size_t total = (size_t)NEXP * NT * KT * 32;
  size_t tid = (size_t)blockIdx.x * blockDim.x + threadIdx.x;
  const size_t gstride = (size_t)gridDim.x * blockDim.x;
  for (; tid < total; tid += gstride) {
    int lane = (int)(tid & 31);
    size_t t2 = tid >> 5;
    int kt = (int)(t2 % KT);
    size_t t3 = t2 / KT;
    int nt = (int)(t3 % NT);
    int e  = (int)(t3 / NT);
    const float* src = W + ((size_t)e * Nrows + (size_t)nt * 16 + (lane & 15)) * Kcols
                         + (size_t)kt * 32 + (lane >> 4) * 8;
    v16bf v;
    #pragma unroll
    for (int i = 0; i < 8; ++i) { v[i] = (__bf16)src[i]; v[8 + i] = (__bf16)src[16 + i]; }
    *(v16bf*)(P + (tid << 4)) = v;
  }
}

// ---------------- kernel 1: router (one wave per token) -----------------
__global__ __launch_bounds__(32) void router_kernel(
    const float* __restrict__ x, const float* __restrict__ gw,
    int* __restrict__ topi, float* __restrict__ topw) {
  const int t = blockIdx.x;
  const int lane = threadIdx.x;
  const float* xr = x + (size_t)t * DDIM;

  float acc[NEXP];
  #pragma unroll
  for (int e = 0; e < NEXP; ++e) acc[e] = 0.f;

  for (int d = lane; d < DDIM; d += 32) {
    float xv = xr[d];
    #pragma unroll
    for (int e = 0; e < NEXP; ++e) acc[e] += xv * gw[e * DDIM + d];
  }
  #pragma unroll
  for (int e = 0; e < NEXP; ++e)
    #pragma unroll
    for (int off = 16; off; off >>= 1) acc[e] += __shfl_xor(acc[e], off, 32);

  if (lane == 0) {
    float mx = acc[0];
    #pragma unroll
    for (int e = 1; e < NEXP; ++e) mx = fmaxf(mx, acc[e]);
    float p[NEXP], s = 0.f;
    #pragma unroll
    for (int e = 0; e < NEXP; ++e) { p[e] = __expf(acc[e] - mx); s += p[e]; }
    float inv = 1.f / s;
    #pragma unroll
    for (int e = 0; e < NEXP; ++e) p[e] *= inv;
    int i0 = 0;
    #pragma unroll
    for (int e = 1; e < NEXP; ++e) if (p[e] > p[i0]) i0 = e;
    int i1 = (i0 == 0) ? 1 : 0;
    #pragma unroll
    for (int e = 0; e < NEXP; ++e) if (e != i0 && p[e] > p[i1]) i1 = e;
    float w0 = p[i0], w1 = p[i1];
    float ws = 1.f / (w0 + w1 + 1e-9f);
    topi[t * 2 + 0] = i0;  topi[t * 2 + 1] = i1;
    topw[t * 2 + 0] = w0 * ws;  topw[t * 2 + 1] = w1 * ws;
  }
}

// ---------------- kernel 2: capacity scan + compaction (1 wave) ---------
__global__ __launch_bounds__(32) void capacity_kernel(
    const int* __restrict__ topi, const float* __restrict__ topw,
    int* __restrict__ cnt, int* __restrict__ tok, float* __restrict__ wgt) {
  const int lane = threadIdx.x;
  const unsigned long long lt = (1ull << lane) - 1ull;
  int c0[NEXP], c1[NEXP], pos[NEXP];
  #pragma unroll
  for (int e = 0; e < NEXP; ++e) { c0[e] = 0; c1[e] = 0; pos[e] = 0; }

  for (int base = 0; base < T_TOK; base += 32) {
    const int t = base + lane;
    const int  i0 = topi[t * 2 + 0], i1 = topi[t * 2 + 1];
    const float w0 = topw[t * 2 + 0], w1 = topw[t * 2 + 1];
    bool keep0 = false, keep1 = false;
    #pragma unroll
    for (int e = 0; e < NEXP; ++e) {
      unsigned long long m0 = __ballot(i0 == e);
      unsigned long long m1 = __ballot(i1 == e);
      if (i0 == e && (c0[e] + (int)__popcll(m0 & lt)) < CAP) keep0 = true;
      if (i1 == e && (c1[e] + (int)__popcll(m1 & lt)) < CAP) keep1 = true;
      c0[e] += (int)__popcll(m0);
      c1[e] += (int)__popcll(m1);
    }
    #pragma unroll
    for (int e = 0; e < NEXP; ++e) {
      bool s0 = keep0 && (i0 == e);
      bool s1 = keep1 && (i1 == e);       // i0 != i1, at most one fires
      bool sel = s0 || s1;
      unsigned long long m = __ballot(sel);
      int idx = pos[e] + (int)__popcll(m & lt);
      if (sel) {
        tok[e * MAXTOK + idx] = t;
        wgt[e * MAXTOK + idx] = s0 ? w0 : w1;
      }
      pos[e] += (int)__popcll(m);
    }
  }
  if (lane < NEXP) cnt[lane] = pos[lane];
}

// ---------------- kernel 3: fused per-expert SwiGLU FFN -----------------
// Block = (expert, 16-token tile), 8 waves. LDS: x-tile bf16 (32KB) +
// h tile bf16 (64KB). N-tiles processed in pairs so each LDS A-fragment
// feeds 4 (phase A) / 2 (phase B) v_wmma_f32_16x16x32_bf16 ops.
template <bool PACKED>
__global__ __launch_bounds__(256) void moe_ffn_kernel(
    const float* __restrict__ x,
    const float* __restrict__ Wg, const float* __restrict__ Wu,
    const float* __restrict__ Wd,
    const __bf16* __restrict__ Pg, const __bf16* __restrict__ Pu,
    const __bf16* __restrict__ Pd,
    const int* __restrict__ cnt, const int* __restrict__ tok,
    const float* __restrict__ wgt, float* __restrict__ out) {
  __shared__ __bf16 xs[16 * DDIM];   // 32 KB
  __shared__ __bf16 hs[16 * HDIM];   // 64 KB
  __shared__ int    tok_sh[16];
  __shared__ float  wgt_sh[16];

  const int e     = blockIdx.x / TILES;
  const int m0    = (blockIdx.x % TILES) * 16;
  const int count = cnt[e];
  if (m0 >= count) return;           // uniform across block

  const int tid  = threadIdx.x;
  const int lane = tid & 31;
  const int wave = tid >> 5;
  const int m    = lane & 15;
  const int hh   = lane >> 4;

  if (tid < 16) {
    int idx = m0 + tid;
    bool v = idx < count;
    tok_sh[tid] = tok[e * MAXTOK + (v ? idx : m0)];
    wgt_sh[tid] = v ? wgt[e * MAXTOK + idx] : 0.f;
  }
  __syncthreads();

  for (int i = tid; i < 16 * DDIM; i += 256) {
    int r = i >> 10, ccol = i & (DDIM - 1);
    xs[i] = (__bf16)x[(size_t)tok_sh[r] * DDIM + ccol];
  }
  __syncthreads();

  // ---- phase A: g = X*Wg^T, u = X*Wu^T, h = silu(g)*u*routeW ----
  {
    const int KT = DDIM / 32;                       // 32 K-steps
    const float*  Fg = Wg + (size_t)e * HDIM * DDIM;
    const float*  Fu = Wu + (size_t)e * HDIM * DDIM;
    const __bf16* Qg = Pg + (size_t)e * (HDIM / 16) * KT * 512;
    const __bf16* Qu = Pu + (size_t)e * (HDIM / 16) * KT * 512;
    #pragma unroll
    for (int p = 0; p < 8; ++p) {                   // 128 N-tiles / 8 waves / 2
      const int nt0 = wave + 16 * p, nt1 = nt0 + 8;
      v8f g0 = {}, u0 = {}, g1 = {}, u1 = {};
      for (int kt = 0; kt < KT; ++kt) {
        const int kb = kt * 32;
        v16bf a = load_a_bf16(xs, m, hh, kb, DDIM);
        v16bf bg0, bu0, bg1, bu1;
        if (PACKED) {
          __builtin_prefetch(Qg + ((size_t)nt0 * KT + kt + 1) * 512, 0, 0);
          bg0 = load_b_packed(Qg, nt0 * KT + kt, lane);
          bu0 = load_b_packed(Qu, nt0 * KT + kt, lane);
          bg1 = load_b_packed(Qg, nt1 * KT + kt, lane);
          bu1 = load_b_packed(Qu, nt1 * KT + kt, lane);
        } else {
          bg0 = load_b_f32(Fg, nt0 * 16, kb, DDIM, lane);
          bu0 = load_b_f32(Fu, nt0 * 16, kb, DDIM, lane);
          bg1 = load_b_f32(Fg, nt1 * 16, kb, DDIM, lane);
          bu1 = load_b_f32(Fu, nt1 * 16, kb, DDIM, lane);
        }
        g0 = __builtin_amdgcn_wmma_f32_16x16x32_bf16(false, a, false, bg0, (short)0, g0, false, false);
        u0 = __builtin_amdgcn_wmma_f32_16x16x32_bf16(false, a, false, bu0, (short)0, u0, false, false);
        g1 = __builtin_amdgcn_wmma_f32_16x16x32_bf16(false, a, false, bg1, (short)0, g1, false, false);
        u1 = __builtin_amdgcn_wmma_f32_16x16x32_bf16(false, a, false, bu1, (short)0, u1, false, false);
      }
      #pragma unroll
      for (int j = 0; j < 8; ++j) {                 // C: M=j+8*hh, N=lane&15
        int row = j + hh * 8;
        float rw = wgt_sh[row];
        float ga = g0[j], ua = u0[j];
        float gb = g1[j], ub = u1[j];
        hs[row * HDIM + nt0 * 16 + m] = (__bf16)((ga / (1.f + __expf(-ga))) * ua * rw);
        hs[row * HDIM + nt1 * 16 + m] = (__bf16)((gb / (1.f + __expf(-gb))) * ub * rw);
      }
    }
  }
  __syncthreads();

  // ---- phase B: out[tok] += h * Wd^T (route weight already folded) ----
  {
    const int KT = HDIM / 32;                       // 64 K-steps
    const float*  Fd = Wd + (size_t)e * DDIM * HDIM;
    const __bf16* Qd = Pd + (size_t)e * (DDIM / 16) * KT * 512;
    #pragma unroll
    for (int p = 0; p < 4; ++p) {                   // 64 N-tiles / 8 waves / 2
      const int nt0 = wave + 16 * p, nt1 = nt0 + 8;
      v8f c0 = {}, c1 = {};
      for (int kt = 0; kt < KT; ++kt) {
        const int kb = kt * 32;
        v16bf a = load_a_bf16(hs, m, hh, kb, HDIM);
        v16bf b0, b1;
        if (PACKED) {
          __builtin_prefetch(Qd + ((size_t)nt0 * KT + kt + 1) * 512, 0, 0);
          b0 = load_b_packed(Qd, nt0 * KT + kt, lane);
          b1 = load_b_packed(Qd, nt1 * KT + kt, lane);
        } else {
          b0 = load_b_f32(Fd, nt0 * 16, kb, HDIM, lane);
          b1 = load_b_f32(Fd, nt1 * 16, kb, HDIM, lane);
        }
        c0 = __builtin_amdgcn_wmma_f32_16x16x32_bf16(false, a, false, b0, (short)0, c0, false, false);
        c1 = __builtin_amdgcn_wmma_f32_16x16x32_bf16(false, a, false, b1, (short)0, c1, false, false);
      }
      #pragma unroll
      for (int j = 0; j < 8; ++j) {
        int row = j + hh * 8;
        size_t o = (size_t)tok_sh[row] * DDIM + m;
        atomicAdd(&out[o + nt0 * 16], c0[j]);
        atomicAdd(&out[o + nt1 * 16], c1[j]);
      }
    }
  }
}

// ---------------- host launcher ----------------------------------------
extern "C" void kernel_launch(void* const* d_in, const int* in_sizes, int n_in,
                              void* d_out, int out_size, void* d_ws, size_t ws_size,
                              hipStream_t stream) {
  const float* x      = (const float*)d_in[0];  // (B,S,D)
  const float* gate_w = (const float*)d_in[1];  // (E,D)
  const float* Wg     = (const float*)d_in[2];  // (E,H,D)
  const float* Wu     = (const float*)d_in[3];  // (E,H,D)
  const float* Wd     = (const float*)d_in[4];  // (E,D,H)
  float* out = (float*)d_out;

  char* ws = (char*)d_ws;
  int*   topi = (int*)(ws);
  float* topw = (float*)(ws + T_TOK * 2 * 4);
  int*   cnt  = (int*)(ws + T_TOK * 4 * 4);
  int*   tok  = (int*)(ws + T_TOK * 4 * 4 + 256);
  float* wgt  = (float*)(ws + T_TOK * 4 * 4 + 256 + NEXP * MAXTOK * 4);

  zero_out_kernel<<<2048, 256, 0, stream>>>(out, (size_t)T_TOK * DDIM);
  router_kernel<<<T_TOK, 32, 0, stream>>>(x, gate_w, topi, topw);
  capacity_kernel<<<1, 32, 0, stream>>>(topi, topw, cnt, tok, wgt);

  if (ws_size >= (size_t)WS_REQUIRED) {
    // bf16 fragment-packed weight path: ~96MB (L2-resident on MI455X)
    __bf16* Pg = (__bf16*)(ws + ROUTE_BYTES);
    __bf16* Pu = Pg + PACK_ELEMS;
    __bf16* Pd = Pu + PACK_ELEMS;
    pack_weights_kernel<<<4096, 256, 0, stream>>>(Wg, Pg, HDIM, DDIM);
    pack_weights_kernel<<<4096, 256, 0, stream>>>(Wu, Pu, HDIM, DDIM);
    pack_weights_kernel<<<4096, 256, 0, stream>>>(Wd, Pd, DDIM, HDIM);
    moe_ffn_kernel<true><<<NEXP * TILES, 256, 0, stream>>>(
        x, Wg, Wu, Wd, Pg, Pu, Pd, cnt, tok, wgt, out);
  } else {
    moe_ffn_kernel<false><<<NEXP * TILES, 256, 0, stream>>>(
        x, Wg, Wu, Wd, nullptr, nullptr, nullptr, cnt, tok, wgt, out);
  }
}
